// Head_58540404244641
// MI455X (gfx1250) — matrix operand order repaired and checked by reading, not compiled, not run on hardware
//
#include <hip/hip_runtime.h>
#include <stdint.h>

// ---------------------------------------------------------------------------
// CDNA5 (gfx1250) single-head attention, bf16 WMMA pipeline, wave32.
//   B=4, T=2048, C=1024, H=128
//   out[t] = softmax_s( (k[t].q[s]) * C^-0.5 ) @ v      (no causal mask)
//
// Pass 0: fp32 -> bf16 conversion (X row-major, W transposed)  [HBM-bound]
// Pass 1: projections K,Q,V via v_wmma_f32_16x16x32_bf16       [matrix-bound]
// Pass 2: flash attention, LDS-staged tiles (async-to-LDS),
//         online softmax, all GEMMs on the WMMA pipe.
// ---------------------------------------------------------------------------

#define BATCH   4
#define SEQ     2048
#define EMB     1024
#define HEADD   128

#if defined(__has_builtin)
#if __has_builtin(__builtin_amdgcn_global_load_async_to_lds_b128)
#define HAVE_ASYNC_LDS 1
#endif
#endif

typedef __attribute__((ext_vector_type(16))) __bf16 v16bf;
typedef __attribute__((ext_vector_type(8)))  float  v8f;

// Exact pointee type the async-to-LDS builtin expects (per hipcc diagnostic):
//   "__attribute__((__vector_size__(4 * sizeof(int)))) int __device__ *"
typedef int v4i_vec __attribute__((vector_size(16)));
typedef __attribute__((address_space(1))) v4i_vec* gvec_ptr;   // global (AS1)
typedef __attribute__((address_space(3))) v4i_vec* lvec_ptr;   // LDS (AS3)

union FragBF { uint32_t u[8]; v16bf v; };
union FragF  { float    f[8]; v8f   v; };

__device__ __forceinline__ uint16_t f2bf(float x) {
    union { float f; uint32_t u; } c; c.f = x;
    uint32_t r = c.u + 0x7FFFu + ((c.u >> 16) & 1u);   // round-to-nearest-even
    return (uint16_t)(r >> 16);
}
__device__ __forceinline__ uint32_t pack_bf(float lo, float hi) {
    return (uint32_t)f2bf(lo) | ((uint32_t)f2bf(hi) << 16);
}
__device__ __forceinline__ v8f wmma_bf16(const FragBF& a, const FragBF& b, v8f c) {
    return __builtin_amdgcn_wmma_f32_16x16x32_bf16(
        false, a.v, false, b.v, (short)0, c, false, false);
}

// ---------------------------------------------------------------------------
// Pass 0a: X [8192,1024] f32 -> bf16 pairs (row-major).
// ---------------------------------------------------------------------------
__global__ __launch_bounds__(256)
void cvt_x_kernel(const float* __restrict__ X, uint32_t* __restrict__ Xb32)
{
    size_t i = (size_t)blockIdx.x * 256 + threadIdx.x;   // 4,194,304 pairs
    const float2 v = *(const float2*)(X + 2 * i);
    Xb32[i] = pack_bf(v.x, v.y);
}

// ---------------------------------------------------------------------------
// Pass 0b: Wk/Wq/Wv [1024,128] f32 -> transposed bf16 pairs Wt [3][128][512].
// ---------------------------------------------------------------------------
__global__ __launch_bounds__(256)
void cvt_w_kernel(const float* __restrict__ Wk, const float* __restrict__ Wq,
                  const float* __restrict__ Wv, uint32_t* __restrict__ Wt32)
{
    int i   = blockIdx.x * 256 + threadIdx.x;            // < 196,608
    int w   = i >> 16;                                   // 65,536 per matrix
    int r   = i & 0xFFFF;
    int col = r >> 9;                                    // 0..127
    int kp  = r & 511;                                   // k pair 0..511
    const float* W = (w == 0) ? Wk : (w == 1) ? Wq : Wv;
    float w0 = W[(size_t)(2 * kp)     * HEADD + col];
    float w1 = W[(size_t)(2 * kp + 1) * HEADD + col];
    Wt32[(((size_t)w * HEADD + col) << 9) + kp] = pack_bf(w0, w1);
}

// ---------------------------------------------------------------------------
// Pass 1: projections.  One wave -> one 16x16 tile. grid=(rows/64, 8, 3).
// Inner loop is pure u32 loads + WMMA (no conversion VALU).
// ---------------------------------------------------------------------------
__global__ __launch_bounds__(128)
void proj_bf16_kernel(const uint32_t* __restrict__ Xb32,
                      const uint32_t* __restrict__ Wt32,
                      const float* __restrict__ bk, const float* __restrict__ bq,
                      const float* __restrict__ bv,
                      uint16_t* __restrict__ Kb, uint16_t* __restrict__ Qb,
                      uint16_t* __restrict__ Vt)
{
    const int w      = blockIdx.z;                       // 0:K 1:Q 2:V
    const float* bias = (w == 0) ? bk : (w == 1) ? bq : bv;

    const int lane  = threadIdx.x & 31;
    const int wid   = threadIdx.x >> 5;
    const int n     = lane & 15;
    const int khalf = lane >> 4;

    const int row0 = (blockIdx.x * 4 + wid) * 16;
    const int col  = blockIdx.y * 16 + n;
    const size_t arow = (size_t)(row0 + n) * 512;        // u32 row base of X
    const size_t wrow = ((size_t)w * HEADD + col) * 512; // u32 row base of Wt

    FragF c;
    #pragma unroll
    for (int r = 0; r < 8; ++r) c.f[r] = 0.f;

    for (int kt = 0; kt < EMB / 2; kt += 16) {           // u32 pairs, 32 k/step
        FragBF a, b;
        #pragma unroll
        for (int v = 0; v < 8; ++v) {
            // A layout: (k>>1) = 8*(v>>2) + 4*khalf + (v&3)
            a.u[v] = Xb32[arow + kt + ((v >> 2) << 3) + (khalf << 2) + (v & 3)];
            // B layout: (k>>1) = 8*(v>>2) + 4*khalf + (v&3)
            b.u[v] = Wt32[wrow + kt + ((v >> 2) << 3) + (khalf << 2) + (v & 3)];
        }
        c.v = wmma_bf16(a, b, c.v);
    }

    const float bcol = bias[col];
    if (w < 2) {
        uint16_t* dst = (w == 0) ? Kb : Qb;
        #pragma unroll
        for (int r = 0; r < 8; ++r) {
            int m = r + 8 * khalf;
            dst[(size_t)(row0 + m) * HEADD + col] = f2bf(c.f[r] + bcol);
        }
    } else {                                             // V stored transposed
        #pragma unroll
        for (int r = 0; r < 8; ++r) {
            int m    = r + 8 * khalf;
            int grow = row0 + m;
            int bi   = grow >> 11;
            int t    = grow & (SEQ - 1);
            Vt[((size_t)bi * HEADD + col) * SEQ + t] = f2bf(c.f[r] + bcol);
        }
    }
}

// ---------------------------------------------------------------------------
// Pass 2: flash attention. grid=(T/64, B), block=128 (4 waves).
// Q/V tiles staged cooperatively into LDS (async-to-LDS where available).
// LDS row strides padded for conflict-free 16-lane fragment reads while
// keeping every 16B chunk aligned for b128 async copies.
// ---------------------------------------------------------------------------
#define QS_STRIDE 68   // u32 per 128-bf16 Q row   (272B, 16B-aligned chunks)
#define VS_STRIDE 20   // u32 per 32-bf16 V row    (80B,  16B-aligned chunks)
#define PB_STRIDE 38   // u16 per P row (19 u32)

__global__ __launch_bounds__(128)
void flash_attn_kernel(const uint32_t* __restrict__ Kb,   // bf16 pairs [B*T,64]
                       const uint32_t* __restrict__ Qb,   // bf16 pairs [B*T,64]
                       const uint32_t* __restrict__ Vt,   // bf16 pairs [B,128,1024]
                       float* __restrict__ out)
{
    __shared__ __align__(16) uint32_t Qs[32 * QS_STRIDE];     // 8704 B
    __shared__ __align__(16) uint32_t Vs[128 * VS_STRIDE];    // 10240 B
    __shared__ __align__(16) uint16_t pbuf[4][16 * PB_STRIDE];// 4864 B

    const int tid   = threadIdx.x;
    const int lane  = tid & 31;
    const int wid   = tid >> 5;
    const int n     = lane & 15;
    const int khalf = lane >> 4;
    const int b     = blockIdx.y;
    const int t0    = blockIdx.x * 64 + wid * 16;
    const size_t rowbase = (size_t)b * SEQ;
    const uint32_t* Vbase = Vt + (size_t)b * HEADD * (SEQ / 2);

    // Resident A-fragments of this wave's 16 "query" rows (the K matrix).
    FragBF aK[4];
    #pragma unroll
    for (int kt = 0; kt < 4; ++kt)
        #pragma unroll
        for (int v = 0; v < 8; ++v) {
            int kh = kt * 16 + ((v >> 2) << 3) + (khalf << 2) + (v & 3); // k>>1
            aK[kt].u[v] = Kb[(rowbase + t0 + n) * 64 + kh];
        }

    float rowM[8], rowL[8];
    FragF acc[8];
    #pragma unroll
    for (int r = 0; r < 8; ++r) { rowM[r] = -3.0e38f; rowL[r] = 0.f; }
    #pragma unroll
    for (int ht = 0; ht < 8; ++ht)
        #pragma unroll
        for (int r = 0; r < 8; ++r) acc[ht].f[r] = 0.f;

    const float scale = 0.03125f;                         // 1024^-0.5

    for (int s0 = 0; s0 < SEQ; s0 += 32) {
        __syncthreads();                                  // prev-iter LDS reads done

        // ---- stage Q (32x128 bf16) and V (128x32 bf16) tiles --------------
        const uint32_t* Qg = Qb + (rowbase + s0) * 64;
        const uint32_t* Vg = Vbase + (s0 >> 1);
        #pragma unroll
        for (int i = 0; i < 4; ++i) {
            int c  = tid + i * 128;                       // chunk id 0..511
            int qr = c >> 4, qq = c & 15;                 // Q: 16 chunks/row
            int vr = c >> 2, vq = c & 3;                  // V: 4 chunks/row
#ifdef HAVE_ASYNC_LDS
            __builtin_amdgcn_global_load_async_to_lds_b128(
                (gvec_ptr)(Qg + qr * 64 + qq * 4),
                (lvec_ptr)(Qs + qr * QS_STRIDE + qq * 4),
                0, 0);
            __builtin_amdgcn_global_load_async_to_lds_b128(
                (gvec_ptr)(Vg + (size_t)vr * (SEQ / 2) + vq * 4),
                (lvec_ptr)(Vs + vr * VS_STRIDE + vq * 4),
                0, 0);
#else
            *(uint4*)(Qs + qr * QS_STRIDE + qq * 4) =
                *(const uint4*)(Qg + qr * 64 + qq * 4);
            *(uint4*)(Vs + vr * VS_STRIDE + vq * 4) =
                *(const uint4*)(Vg + (size_t)vr * (SEQ / 2) + vq * 4);
#endif
        }
        if (s0 + 32 < SEQ)                                // prefetch next Q tile
            __builtin_prefetch((const void*)(Qb + (rowbase + s0 + 32 + n) * 64), 0, 0);
#ifdef HAVE_ASYNC_LDS
#if __has_builtin(__builtin_amdgcn_s_wait_asynccnt)
        __builtin_amdgcn_s_wait_asynccnt(0);
#else
        asm volatile("s_wait_asynccnt 0x0" ::: "memory");
#endif
#endif
        __syncthreads();                                  // tiles visible

        // ---- scores: two 16-wide column tiles, K dim = 128 ----------------
        FragF sc0, sc1;
        #pragma unroll
        for (int r = 0; r < 8; ++r) { sc0.f[r] = 0.f; sc1.f[r] = 0.f; }
        #pragma unroll
        for (int kt = 0; kt < 4; ++kt) {
            FragBF bq0, bq1;
            #pragma unroll
            for (int v = 0; v < 8; ++v) {
                int kh = kt * 16 + ((v >> 2) << 3) + (khalf << 2) + (v & 3); // B layout, k>>1
                bq0.u[v] = Qs[(n)      * QS_STRIDE + kh];
                bq1.u[v] = Qs[(16 + n) * QS_STRIDE + kh];
            }
            sc0.v = wmma_bf16(aK[kt], bq0, sc0.v);
            sc1.v = wmma_bf16(aK[kt], bq1, sc1.v);
        }

        // ---- online softmax (rows live in 16-lane halves) -----------------
        #pragma unroll
        for (int r = 0; r < 8; ++r) {
            float x0 = sc0.f[r] * scale, x1 = sc1.f[r] * scale;
            float mx = fmaxf(x0, x1);
            #pragma unroll
            for (int m = 1; m < 16; m <<= 1) mx = fmaxf(mx, __shfl_xor(mx, m, 32));
            float Mn    = fmaxf(rowM[r], mx);
            float alpha = __expf(rowM[r] - Mn);
            float p0    = __expf(x0 - Mn);
            float p1    = __expf(x1 - Mn);
            float sum   = p0 + p1;
            #pragma unroll
            for (int m = 1; m < 16; m <<= 1) sum += __shfl_xor(sum, m, 32);
            rowL[r] = rowL[r] * alpha + sum;
            rowM[r] = Mn;
            #pragma unroll
            for (int ht = 0; ht < 8; ++ht) acc[ht].f[r] *= alpha;
            sc0.f[r] = p0; sc1.f[r] = p1;
        }

        // ---- transpose P (C layout -> A layout) through per-wave LDS ------
        {
            uint16_t* pb = pbuf[wid];
            #pragma unroll
            for (int r = 0; r < 8; ++r) {
                int m = r + 8 * khalf;
                pb[m * PB_STRIDE + n]      = f2bf(sc0.f[r]);
                pb[m * PB_STRIDE + 16 + n] = f2bf(sc1.f[r]);
            }
        }
        __syncthreads();
        FragBF aP;
        {
            const uint32_t* pb32 = (const uint32_t*)pbuf[wid];
            #pragma unroll
            for (int v = 0; v < 8; ++v) {
                int kh = ((v >> 2) << 3) + (khalf << 2) + (v & 3); // A layout k>>1
                aP.u[v] = pb32[n * (PB_STRIDE / 2) + kh];
            }
        }

        // ---- out += P @ V, 8 column tiles of 16 ---------------------------
        #pragma unroll
        for (int ht = 0; ht < 8; ++ht) {
            FragBF bv_;
            #pragma unroll
            for (int v = 0; v < 8; ++v)
                bv_.u[v] = Vs[(ht * 16 + n) * VS_STRIDE + ((v >> 2) << 3) + (khalf << 2) + (v & 3)];
            acc[ht].v = wmma_bf16(aP, bv_, acc[ht].v);
        }
    }

    // ---- normalize + store -------------------------------------------------
    #pragma unroll
    for (int ht = 0; ht < 8; ++ht) {
        #pragma unroll
        for (int r = 0; r < 8; ++r) {
            int m = r + 8 * khalf;
            out[(rowbase + t0 + m) * HEADD + ht * 16 + n] = acc[ht].f[r] / rowL[r];
        }
    }
}

// ---------------------------------------------------------------------------
extern "C" void kernel_launch(void* const* d_in, const int* in_sizes, int n_in,
                              void* d_out, int out_size, void* d_ws, size_t ws_size,
                              hipStream_t stream) {
    const float* X  = (const float*)d_in[0];
    const float* Wk = (const float*)d_in[1];
    const float* bk = (const float*)d_in[2];
    const float* Wq = (const float*)d_in[3];
    const float* bq = (const float*)d_in[4];
    const float* Wv = (const float*)d_in[5];
    const float* bv = (const float*)d_in[6];
    float* out = (float*)d_out;

    // Workspace layout (u16 units): Xb | Wt | K | Q | Vt   (~24 MB total)
    uint16_t* w16 = (uint16_t*)d_ws;
    uint16_t* Xb  = w16;                                  // 8192*1024
    uint16_t* Wt  = Xb + (size_t)BATCH * SEQ * EMB;       // 3*128*1024
    uint16_t* Kb  = Wt + (size_t)3 * HEADD * EMB;         // 8192*128
    uint16_t* Qb  = Kb + (size_t)BATCH * SEQ * HEADD;
    uint16_t* Vt  = Qb + (size_t)BATCH * SEQ * HEADD;

    cvt_x_kernel<<<BATCH * SEQ * EMB / 2 / 256, 256, 0, stream>>>(X, (uint32_t*)Xb);
    cvt_w_kernel<<<3 * HEADD * (EMB / 2) / 256, 256, 0, stream>>>(Wk, Wq, Wv, (uint32_t*)Wt);

    dim3 g1(BATCH * SEQ / 64, HEADD / 16, 3);             // (128, 8, 3)
    proj_bf16_kernel<<<g1, 128, 0, stream>>>((const uint32_t*)Xb, (const uint32_t*)Wt,
                                             bk, bq, bv, Kb, Qb, Vt);

    dim3 g2(SEQ / 64, BATCH);                             // (32, 4)
    flash_attn_kernel<<<g2, 128, 0, stream>>>((const uint32_t*)Kb,
                                              (const uint32_t*)Qb,
                                              (const uint32_t*)Vt, out);
}
